// Block_65386582114509
// MI455X (gfx1250) — matrix operand order repaired and checked by reading, not compiled, not run on hardware
//
#include <hip/hip_runtime.h>
#include <cstdint>

#define DMODEL 1024
#define SLEN   1024
#define SENC   512
#define NHEAD  16
#define HEADD  64
#define BATCH  8
#define NEGV  (-100000.0f)

typedef __attribute__((ext_vector_type(16))) __bf16 v16bf;
typedef __attribute__((ext_vector_type(8)))  float  v8f;
typedef int v4i_ __attribute__((ext_vector_type(4)));

union FragAB { v16bf v; uint4 u[2]; };

#if __has_builtin(__builtin_amdgcn_global_load_async_to_lds_b128)
#define ASYNC_LDS 1
#else
#define ASYNC_LDS 0
#endif

// Stage 16 bytes global -> LDS. Async DMA path on CDNA5 (ASYNCcnt-tracked),
// falls back to register staging if the builtin is unavailable.
__device__ __forceinline__ void stage16(const unsigned short* g, unsigned short* l) {
#if ASYNC_LDS
  __builtin_amdgcn_global_load_async_to_lds_b128(
      (__attribute__((address_space(1))) v4i_*)(g),
      (__attribute__((address_space(3))) v4i_*)(l), 0, 0);
#else
  *(uint4*)l = *(const uint4*)g;
#endif
}

__device__ __forceinline__ void stage_wait() {
#if ASYNC_LDS
#if __has_builtin(__builtin_amdgcn_s_wait_asynccnt)
  __builtin_amdgcn_s_wait_asynccnt(0);
#else
  asm volatile("s_wait_asynccnt 0x0" ::: "memory");
#endif
#endif
}

__device__ __forceinline__ unsigned short f2bf(float x) {
  union { float f; unsigned int u; } p; p.f = x;
  unsigned int u = p.u;
  unsigned int r = u + 0x7FFFu + ((u >> 16) & 1u);   // round-to-nearest-even
  return (unsigned short)(r >> 16);
}

__device__ __forceinline__ float gelu_tanh(float x) {
  float x3 = x * x * x;
  float t  = tanhf(0.7978845608028654f * (x + 0.044715f * x3));
  return 0.5f * x * (1.0f + t);
}

// ---------------- f32 -> bf16 convert ----------------
__global__ void cvt_kernel(const float* __restrict__ in,
                           unsigned short* __restrict__ out, long n) {
  long i = (long)blockIdx.x * blockDim.x + threadIdx.x;
  if (i < n) out[i] = f2bf(in[i]);
}

// ---------------- generic bf16 WMMA GEMM (double-buffered) ----------------
// C[M,N] = A[M,K] * B[K,N] + bias ; block tile 128x128, K-step 32.
// 8 waves, each wave computes 64x32 (4x2 WMMA tiles). A tile staged with
// async global->LDS DMA; B tile transposed into LDS ([n][k]) via registers.
template <bool GELU, bool OUTBF16>
__global__ __launch_bounds__(256) void gemm_kernel(
    const unsigned short* __restrict__ A, int lda,
    const unsigned short* __restrict__ Bm, int ldb,
    const float* __restrict__ bias,
    float* __restrict__ Cf, unsigned short* __restrict__ Cb, int ldc, int K) {
  __shared__ unsigned short Ash[2][128 * 40];   // [m][k], stride 40 (pad 8)
  __shared__ unsigned short Bsh[2][128 * 40];   // [n][k], stride 40 (pad 8)

  const int tid  = threadIdx.x;
  const int wave = tid >> 5, lane = tid & 31;
  const int hh = lane >> 4, cc = lane & 15;
  const int waveM = (wave >> 2) * 64;   // 0 / 64
  const int waveN = (wave & 3) * 32;    // 0..96
  const long rowBase = (long)blockIdx.x * 128;
  const long colBase = (long)blockIdx.y * 128;

  v8f zero8 = {0.f, 0.f, 0.f, 0.f, 0.f, 0.f, 0.f, 0.f};
  v8f acc[4][2];
#pragma unroll
  for (int i = 0; i < 4; i++)
#pragma unroll
    for (int j = 0; j < 2; j++) acc[i][j] = zero8;

  const int aRow = tid >> 1, aChunk = (tid & 1) * 16;
  const int bRow = tid >> 3, bCol = (tid & 7) * 16;
  const unsigned short* Abase = A + (rowBase + aRow) * (long)lda + aChunk;
  const unsigned short* Bbase = Bm + (long)bRow * ldb + colBase + bCol;

  const int nK = K >> 5;

  // ---- prologue: stage tile 0 into buffer 0 ----
  {
    const unsigned short* g = Abase;
    unsigned short* l = &Ash[0][aRow * 40 + aChunk];
    stage16(g, l);
    stage16(g + 8, l + 8);
    const uint4* bp = (const uint4*)Bbase;
    union { uint4 u[2]; unsigned short h[16]; } bt;
    bt.u[0] = bp[0]; bt.u[1] = bp[1];
#pragma unroll
    for (int j = 0; j < 16; j++) Bsh[0][(bCol + j) * 40 + bRow] = bt.h[j];
  }
  stage_wait();
  __syncthreads();

  for (int t = 0; t < nK; t++) {
    const int buf = t & 1, nbuf = buf ^ 1;
    uint4 b0, b1;
    const bool more = (t + 1) < nK;
    if (more) {
      // prefetch next tile: async A, global->reg B
      const int kt = (t + 1) << 5;
      const unsigned short* g = Abase + kt;
      unsigned short* l = &Ash[nbuf][aRow * 40 + aChunk];
      stage16(g, l);
      stage16(g + 8, l + 8);
      const uint4* bp = (const uint4*)(Bbase + (long)kt * ldb);
      b0 = bp[0]; b1 = bp[1];
    }

    // ---- compute on current buffer ----
    FragAB bfr[2];
#pragma unroll
    for (int ni = 0; ni < 2; ni++) {
      int n = waveN + ni * 16 + cc;
      bfr[ni].u[0] = *(const uint4*)&Bsh[buf][n * 40 + hh * 8];
      bfr[ni].u[1] = *(const uint4*)&Bsh[buf][n * 40 + 16 + hh * 8];
    }
#pragma unroll
    for (int mi = 0; mi < 4; mi++) {
      int m = waveM + mi * 16 + cc;
      FragAB af;
      af.u[0] = *(const uint4*)&Ash[buf][m * 40 + hh * 8];
      af.u[1] = *(const uint4*)&Ash[buf][m * 40 + 16 + hh * 8];
#pragma unroll
      for (int ni = 0; ni < 2; ni++) {
        acc[mi][ni] = __builtin_amdgcn_wmma_f32_16x16x32_bf16(
            false, af.v, false, bfr[ni].v, (short)0, acc[mi][ni], false, false);
      }
    }

    if (more) {
      union { uint4 u[2]; unsigned short h[16]; } bt;
      bt.u[0] = b0; bt.u[1] = b1;
#pragma unroll
      for (int j = 0; j < 16; j++) Bsh[nbuf][(bCol + j) * 40 + bRow] = bt.h[j];
    }
    stage_wait();
    __syncthreads();
  }

  // ---- epilogue: bias (+GELU), store f32 or bf16 ----
#pragma unroll
  for (int ni = 0; ni < 2; ni++) {
    long col = colBase + waveN + ni * 16 + cc;
    float bv = bias ? bias[col] : 0.f;
#pragma unroll
    for (int mi = 0; mi < 4; mi++) {
#pragma unroll
      for (int e = 0; e < 8; e++) {
        long row = rowBase + waveM + mi * 16 + e + hh * 8;
        float v = acc[mi][ni][e] + bv;
        if (GELU) v = gelu_tanh(v);
        if (OUTBF16) Cb[row * (long)ldc + col] = f2bf(v);
        else         Cf[row * (long)ldc + col] = v;
      }
    }
  }
}

// ---------------- flash-style WMMA attention ----------------
// Block: 8 waves, 128 query rows per block; each wave owns 16 rows so softmax
// statistics are wave-local (shfl_xor reductions within 16-lane halves).
template <bool CAUSAL>
__global__ __launch_bounds__(256) void attn_kernel(
    const unsigned short* __restrict__ Qsrc,   // qkv bf16, stride 3072
    const unsigned short* __restrict__ KVsrc,  // qkv (self) or cross-kv
    int kvStride, int kOff, int vOff, int Skv,
    const int* __restrict__ padmask,           // [B,SE], cross only
    unsigned short* __restrict__ out) {        // [B*S, D] bf16 merged heads
  __shared__ unsigned short Ksh[32 * 72];      // [key][d]
  __shared__ unsigned short Vsh[64 * 40];      // [d][key] (transposed)
  __shared__ unsigned short Psh[8 * 16 * 40];  // per-wave P tile
  __shared__ float amask[SENC];

  const int tid  = threadIdx.x;
  const int wave = tid >> 5, lane = tid & 31;
  const int hh = lane >> 4, cc = lane & 15;
  const int qb = blockIdx.x, head = blockIdx.y, b = blockIdx.z;
  const int  qRow0 = qb * 128 + wave * 16;
  const long qRowG = (long)b * SLEN + qRow0;

  if (!CAUSAL) {
    for (int i = tid; i < Skv; i += 256)
      amask[i] = (float)padmask[b * SENC + i] * NEGV;
  }

  // Q fragments (HD=64 -> 2 K-chunks of 32), loaded once from global
  FragAB qa[2];
  {
    const unsigned short* Qb = Qsrc + (qRowG + cc) * 3072 + head * HEADD;
#pragma unroll
    for (int ks = 0; ks < 2; ks++) {
      qa[ks].u[0] = *(const uint4*)(Qb + ks * 32 + hh * 8);
      qa[ks].u[1] = *(const uint4*)(Qb + ks * 32 + 16 + hh * 8);
    }
  }

  float mrow[8], lrow[8];
  v8f zero8 = {0.f, 0.f, 0.f, 0.f, 0.f, 0.f, 0.f, 0.f};
  v8f o[4];
#pragma unroll
  for (int e = 0; e < 8; e++) { mrow[e] = -3.0e38f; lrow[e] = 0.f; }
#pragma unroll
  for (int i = 0; i < 4; i++) o[i] = zero8;

  unsigned short* Pw = Psh + wave * 16 * 40;
  const int ldK = tid >> 3;          // key row 0..31
  const int ldD = (tid & 7) * 8;     // d offset
  const unsigned short* Kbase = KVsrc + kOff + head * HEADD;
  const unsigned short* Vbase = KVsrc + vOff + head * HEADD;
  const long kvRowB = (long)b * Skv;
  const int nTiles = CAUSAL ? (qb + 1) * 4 : (Skv / 32);
  const float inv = 0.125f;   // 1/sqrt(64)

  for (int t = 0; t < nTiles; t++) {
    const int keyBase = t * 32;
    __syncthreads();
    {
      // K tile: async global->LDS DMA
      const unsigned short* kr = Kbase + (kvRowB + keyBase + ldK) * (long)kvStride + ldD;
      stage16(kr, &Ksh[ldK * 72 + ldD]);
      // V tile: transpose into [d][key] via registers
      const unsigned short* vr = Vbase + (kvRowB + keyBase + ldK) * (long)kvStride + ldD;
      union { uint4 u; unsigned short h[8]; } vt;
      vt.u = *(const uint4*)vr;
#pragma unroll
      for (int j = 0; j < 8; j++) Vsh[(ldD + j) * 40 + ldK] = vt.h[j];
    }
    stage_wait();
    __syncthreads();

    // scores: S = Q(16x64) * K^T  -> two 16x16 C tiles
    v8f sc0 = zero8, sc1 = zero8;
#pragma unroll
    for (int ks = 0; ks < 2; ks++) {
      FragAB kb;
      kb.u[0] = *(const uint4*)&Ksh[cc * 72 + ks * 32 + hh * 8];
      kb.u[1] = *(const uint4*)&Ksh[cc * 72 + ks * 32 + 16 + hh * 8];
      sc0 = __builtin_amdgcn_wmma_f32_16x16x32_bf16(false, qa[ks].v, false, kb.v,
                                                    (short)0, sc0, false, false);
      kb.u[0] = *(const uint4*)&Ksh[(16 + cc) * 72 + ks * 32 + hh * 8];
      kb.u[1] = *(const uint4*)&Ksh[(16 + cc) * 72 + ks * 32 + 16 + hh * 8];
      sc1 = __builtin_amdgcn_wmma_f32_16x16x32_bf16(false, qa[ks].v, false, kb.v,
                                                    (short)0, sc1, false, false);
    }

    // online softmax (per-row stats; rows live at elem e + 8*hh)
#pragma unroll
    for (int e = 0; e < 8; e++) {
      int qg = qRow0 + e + hh * 8;
      float s0 = sc0[e] * inv, s1 = sc1[e] * inv;
      int k0 = keyBase + cc, k1 = keyBase + 16 + cc;
      if (CAUSAL) {
        s0 = (k0 > qg) ? NEGV : s0;
        s1 = (k1 > qg) ? NEGV : s1;
      } else {
        s0 += amask[k0];
        s1 += amask[k1];
      }
      float tm = fmaxf(s0, s1);
      tm = fmaxf(tm, __shfl_xor(tm, 1, 32));
      tm = fmaxf(tm, __shfl_xor(tm, 2, 32));
      tm = fmaxf(tm, __shfl_xor(tm, 4, 32));
      tm = fmaxf(tm, __shfl_xor(tm, 8, 32));
      float mnew  = fmaxf(mrow[e], tm);
      float alpha = __expf(mrow[e] - mnew);
      mrow[e] = mnew;
      float p0 = __expf(s0 - mnew);
      float p1 = __expf(s1 - mnew);
      float rs = p0 + p1;
      rs += __shfl_xor(rs, 1, 32);
      rs += __shfl_xor(rs, 2, 32);
      rs += __shfl_xor(rs, 4, 32);
      rs += __shfl_xor(rs, 8, 32);
      lrow[e] = lrow[e] * alpha + rs;
      o[0][e] *= alpha; o[1][e] *= alpha; o[2][e] *= alpha; o[3][e] *= alpha;
      int prow = e + hh * 8;
      Pw[prow * 40 + cc]      = f2bf(p0);
      Pw[prow * 40 + 16 + cc] = f2bf(p1);
    }
    __syncthreads();

    // O += P(16x32) * V(32x64)
    FragAB pa;
    pa.u[0] = *(const uint4*)&Pw[cc * 40 + hh * 8];
    pa.u[1] = *(const uint4*)&Pw[cc * 40 + 16 + hh * 8];
#pragma unroll
    for (int ont = 0; ont < 4; ont++) {
      FragAB vb;
      vb.u[0] = *(const uint4*)&Vsh[(ont * 16 + cc) * 40 + hh * 8];
      vb.u[1] = *(const uint4*)&Vsh[(ont * 16 + cc) * 40 + 16 + hh * 8];
      o[ont] = __builtin_amdgcn_wmma_f32_16x16x32_bf16(false, pa.v, false, vb.v,
                                                       (short)0, o[ont], false, false);
    }
  }

  // normalize + merged-heads bf16 store
#pragma unroll
  for (int ont = 0; ont < 4; ont++) {
    int col = head * HEADD + ont * 16 + cc;
#pragma unroll
    for (int e = 0; e < 8; e++) {
      long row = qRowG + e + hh * 8;
      out[row * DMODEL + col] = f2bf(o[ont][e] / lrow[e]);
    }
  }
}

// ---------------- fused residual-add + LayerNorm ----------------
// v = A + cs*(B [+ C]) ; out = LN(v)*g + b ; writes f32 and/or bf16.
__global__ __launch_bounds__(256) void ln_fused_kernel(
    const float* __restrict__ A, const float* __restrict__ Bv,
    const float* __restrict__ Cv, float cs,
    const float* __restrict__ g, const float* __restrict__ bb,
    float* __restrict__ outF, unsigned short* __restrict__ outB) {
  __shared__ float r1[256], r2[256];
  const int row = blockIdx.x;
  const long base = (long)row * DMODEL;
  const int t = threadIdx.x;
  float v[4], s = 0.f, s2 = 0.f;
#pragma unroll
  for (int i = 0; i < 4; i++) {
    int idx = t + i * 256;
    float x = A[base + idx];
    float u = Bv[base + idx];
    if (Cv) u += Cv[base + idx];
    x += cs * u;
    v[i] = x; s += x; s2 += x * x;
  }
  r1[t] = s; r2[t] = s2;
  __syncthreads();
  for (int offd = 128; offd > 0; offd >>= 1) {
    if (t < offd) { r1[t] += r1[t + offd]; r2[t] += r2[t + offd]; }
    __syncthreads();
  }
  float mean = r1[0] * (1.0f / DMODEL);
  float var  = r2[0] * (1.0f / DMODEL) - mean * mean;
  float rstd = rsqrtf(var + 1e-5f);
#pragma unroll
  for (int i = 0; i < 4; i++) {
    int idx = t + i * 256;
    float y = (v[i] - mean) * rstd * g[idx] + bb[idx];
    if (outF) outF[base + idx] = y;
    if (outB) outB[base + idx] = f2bf(y);
  }
}

// ---------------- host ----------------
extern "C" void kernel_launch(void* const* d_in, const int* in_sizes, int n_in,
                              void* d_out, int out_size, void* d_ws, size_t ws_size,
                              hipStream_t stream) {
  (void)in_sizes; (void)n_in; (void)out_size; (void)ws_size;
  const float* x       = (const float*)d_in[0];
  const float* enc     = (const float*)d_in[1];
  const int*   pmask   = (const int*)d_in[2];
  const float* w_attn  = (const float*)d_in[3];
  const float* b_attn  = (const float*)d_in[4];
  const float* w_oproj = (const float*)d_in[5];
  const float* b_oproj = (const float*)d_in[6];
  const float* w_fc    = (const float*)d_in[7];
  const float* b_fc    = (const float*)d_in[8];
  const float* w_mproj = (const float*)d_in[9];
  const float* b_mproj = (const float*)d_in[10];
  const float* ln1_g   = (const float*)d_in[11];
  const float* ln1_b   = (const float*)d_in[12];
  const float* ln2_g   = (const float*)d_in[13];
  const float* ln2_b   = (const float*)d_in[14];
  float* out = (float*)d_out;

  char* ws = (char*)d_ws;
  size_t off = 0;
  auto alloc = [&](size_t bytes) -> void* {
    void* p = ws + off;
    off = (off + bytes + 255) & ~(size_t)255;
    return p;
  };
  const long nWA = 1024L * 3072, nWO = 1024L * 1024, nWF = 1024L * 4096, nWM = 4096L * 1024;
  const long nX  = 8L * 1024 * 1024, nE = 8L * 512 * 1024;
  const long nQKV = 8192L * 3072, nKVC = 4096L * 2048;
  const long nBSD = 8192L * 1024, nFC = 8192L * 4096;

  unsigned short* wbA  = (unsigned short*)alloc(nWA * 2);
  unsigned short* wbO  = (unsigned short*)alloc(nWO * 2);
  unsigned short* wbF  = (unsigned short*)alloc(nWF * 2);
  unsigned short* wbM  = (unsigned short*)alloc(nWM * 2);
  unsigned short* xb   = (unsigned short*)alloc(nX * 2);
  unsigned short* encb = (unsigned short*)alloc(nE * 2);
  unsigned short* qkvb = (unsigned short*)alloc(nQKV * 2);
  unsigned short* kvcb = (unsigned short*)alloc(nKVC * 2);
  unsigned short* attS = (unsigned short*)alloc(nBSD * 2);
  unsigned short* attC = (unsigned short*)alloc(nBSD * 2);
  float*          oS   = (float*)alloc(nBSD * 4);
  float*          oC   = (float*)alloc(nBSD * 4);
  float*          nF   = (float*)alloc(nBSD * 4);
  unsigned short* nB   = (unsigned short*)alloc(nBSD * 2);
  unsigned short* fcB  = (unsigned short*)alloc(nFC * 2);
  float*          mF   = (float*)alloc(nBSD * 4);

  auto cvt = [&](const float* src, unsigned short* dst, long n) {
    cvt_kernel<<<(unsigned)((n + 255) / 256), 256, 0, stream>>>(src, dst, n);
  };
  cvt(w_attn, wbA, nWA);
  cvt(w_oproj, wbO, nWO);
  cvt(w_fc, wbF, nWF);
  cvt(w_mproj, wbM, nWM);
  cvt(x, xb, nX);
  cvt(enc, encb, nE);

  // QKV = x @ w_attn + b_attn  -> bf16
  gemm_kernel<false, true><<<dim3(64, 24), 256, 0, stream>>>(
      xb, 1024, wbA, 3072, b_attn, nullptr, qkvb, 3072, 1024);
  // cross KV = enc @ w_attn[:, D:] + b_attn[D:]
  gemm_kernel<false, true><<<dim3(32, 16), 256, 0, stream>>>(
      encb, 1024, wbA + 1024, 3072, b_attn + 1024, nullptr, kvcb, 2048, 1024);

  // self attention (causal) and cross attention (padding mask)
  attn_kernel<true><<<dim3(8, 16, 8), 256, 0, stream>>>(
      qkvb, qkvb, 3072, 1024, 2048, 1024, nullptr, attS);
  attn_kernel<false><<<dim3(8, 16, 8), 256, 0, stream>>>(
      qkvb, kvcb, 2048, 0, 1024, 512, pmask, attC);

  // output projections
  gemm_kernel<false, false><<<dim3(64, 8), 256, 0, stream>>>(
      attS, 1024, wbO, 1024, b_oproj, oS, nullptr, 1024, 1024);
  gemm_kernel<false, false><<<dim3(64, 8), 256, 0, stream>>>(
      attC, 1024, wbO, 1024, b_oproj, oC, nullptr, 1024, 1024);

  // n = LN1(x + 0.5*(oS + oC)) -> f32 + bf16
  ln_fused_kernel<<<8192, 256, 0, stream>>>(x, oS, oC, 0.5f, ln1_g, ln1_b, nF, nB);

  // fc = gelu(n @ w_fc + b_fc) -> bf16
  gemm_kernel<true, true><<<dim3(64, 32), 256, 0, stream>>>(
      nB, 1024, wbF, 4096, b_fc, nullptr, fcB, 4096, 1024);
  // m = fc @ w_mproj + b_mproj -> f32
  gemm_kernel<false, false><<<dim3(64, 8), 256, 0, stream>>>(
      fcB, 4096, wbM, 1024, b_mproj, mF, nullptr, 1024, 4096);

  // out = LN2(n + m)
  ln_fused_kernel<<<8192, 256, 0, stream>>>(nF, mF, nullptr, 1.0f, ln2_g, ln2_b, out, nullptr);
}
